// Attention_28552942584284
// MI455X (gfx1250) — matrix operand order repaired and checked by reading, not compiled, not run on hardware
//
#include <hip/hip_runtime.h>

// Attention: scores = exp(QK^T)/sqrt(d); masked_fill; softmax; out = P@V, summed over q.
// B=16, SQ=1024, SKV=1024, D=256.  Outputs: [B*D summed output | B*SQ*SKV p_attn].
//
// gfx1250 paths used: v_wmma_f32_16x16x32_f16 (QK^T), tensor_load_to_lds (TDM DMA of
// the Q tile with hardware LDS padding), s_wait_tensorcnt, global_prefetch_b8.

typedef __attribute__((ext_vector_type(16))) _Float16     v16h;
typedef __attribute__((ext_vector_type(8)))  float        v8f;
typedef __attribute__((ext_vector_type(4)))  unsigned int u32x4;
typedef __attribute__((ext_vector_type(8)))  unsigned int u32x8;

namespace {
constexpr int kB   = 16;
constexpr int kSQ  = 1024;
constexpr int kSKV = 1024;
constexpr int kD   = 256;
constexpr int kQTile = 16;              // q rows per workgroup
constexpr int kThreads = 128;           // 4 wave32
constexpr float kInvSqrtD = 0.0625f;    // 1/sqrt(256)

// Q tile in LDS as f32, padded: row pitch 260 dwords -> %64 = 4 (conflict-free frag loads).
// The TDM descriptor reproduces this pitch via pad_enable (4 dwords every 256 dwords).
constexpr int kQPitch = kD + 4;         // 260 floats
constexpr int kEPad   = kSKV + 6;       // half stride 1030 -> 515 dwords, %64 = 3 (odd)
}

__global__ void __launch_bounds__(256) zero_outsum_kernel(float* out) {
    out[blockIdx.x * 256 + threadIdx.x] = 0.0f;
}

__global__ void __launch_bounds__(kThreads) attention_tile_kernel(
    const float* __restrict__ Q,
    const float* __restrict__ K,
    const float* __restrict__ V,
    const int*   __restrict__ mask,
    float* __restrict__ outSum,     // [B, D] (pre-zeroed; atomic accumulation)
    float* __restrict__ pOut)       // [B, SQ, SKV]
{
    __shared__ float    sQ[kQTile][kQPitch];
    __shared__ _Float16 sE[kQTile][kEPad];
    __shared__ float    sW[kSKV];
    __shared__ float    sRowPart[kQTile][8];
    __shared__ float    sRinv[kQTile];

    const int tid  = threadIdx.x;
    const int lane = tid & 31;
    const int wave = tid >> 5;
    const int b    = blockIdx.y;
    const int q0   = blockIdx.x * kQTile;

    // ---- TDM: DMA the 16x256 f32 Q tile into LDS with hardware row padding ----
    if (wave == 0) {
        const unsigned long long ga =
            (unsigned long long)(uintptr_t)(Q + ((size_t)b * kSQ + q0) * kD);
        const unsigned int lds = (unsigned int)(uintptr_t)&sQ[0][0]; // flat->LDS: low 32 bits

        // D# group 0: count=1 | lds_addr | global_addr[56:0] | type=2 ("image")
        u32x4 g0;
        g0[0] = 1u;                                     // count=1, user descriptor
        g0[1] = lds;                                    // lds_addr (bytes)
        g0[2] = (unsigned int)ga;                       // global_addr[31:0]
        g0[3] = (unsigned int)(ga >> 32) | (2u << 30);  // global_addr[56:32] | type=2

        // D# group 1: data_size=4B (2), pad_enable, pad_interval=256dw (7), pad_amount=4dw (3)
        u32x8 g1;
        g1[0] = (2u << 16) | (1u << 20) | (7u << 22) | (3u << 25);
        g1[1] = ((unsigned int)kD & 0xFFFFu) << 16;     // tensor_dim0 = 256 (low 16 in [63:48])
        g1[2] = ((unsigned int)kSQ & 0xFFFFu) << 16;    // tensor_dim1 = 1024 (low 16 in [95:80])
        g1[3] = ((unsigned int)kD) << 16;               // tile_dim0 = 256 in [127:112]
        g1[4] = (unsigned int)kQTile;                   // tile_dim1 = 16 in [143:128]
        g1[5] = (unsigned int)kD;                       // tensor_dim0_stride = 256 in [191:160]
        g1[6] = 0u;
        g1[7] = 0u;

        // 2D tensor: VADDR2/VADDR3 omitted (null).
        asm volatile("tensor_load_to_lds %0, %1" :: "s"(g0), "s"(g1) : "memory");
        __builtin_amdgcn_s_wait_tensorcnt(0);
    }
    __syncthreads();

    // ---- Load per-wave WMMA A-fragments (Q, 16x32 f16 per 32-wide D chunk) ----
    // A layout (matches B): lane L holds row M=L&15; lanes 0-15 carry K=0..15,
    // lanes 16-31 carry K=16..31 within the 32-wide chunk.
    const int arow = lane & 15;
    const int alg  = (lane >> 4) * 16;
    v16h aF[kD / 32];
    #pragma unroll
    for (int dc = 0; dc < kD / 32; ++dc) {
        const float* qp = &sQ[arow][dc * 32];
        #pragma unroll
        for (int h = 0; h < 8; ++h) {
            aF[dc][h]     = (_Float16)qp[alg + h];
            aF[dc][8 + h] = (_Float16)qp[alg + 8 + h];
        }
    }

    // ---- Phase A: S = Q K^T per 16x16 kv tile via v_wmma_f32_16x16x32_f16 ----
    // B layout: lane holds column N=lane&15 (a kv row); lanes 0-15 carry K=0..15,
    // lanes 16-31 carry K=16..31 of the 32-wide D chunk.
    const int bn     = lane & 15;
    const int bkbase = (lane >> 4) * 16;
    const int mrow   = (lane >> 4) * 8;

    for (int t = wave; t < kSKV / 16; t += 4) {
        const int kv0 = t * 16;
        const float* krow = K + ((size_t)b * kSKV + kv0 + bn) * kD + bkbase;
        // prefetch next tile's slice of K (global_prefetch_b8)
        if (t + 4 < kSKV / 16)
            __builtin_prefetch(krow + 4 * 16 * kD, 0, 1);

        v8f acc = {};
        #pragma unroll
        for (int dc = 0; dc < kD / 32; ++dc) {
            const float4* kp4 = reinterpret_cast<const float4*>(krow + dc * 32);
            float4 x0 = kp4[0], x1 = kp4[1], x2 = kp4[2], x3 = kp4[3];
            v16h bF;
            bF[0]  = (_Float16)x0.x; bF[1]  = (_Float16)x0.y;
            bF[2]  = (_Float16)x0.z; bF[3]  = (_Float16)x0.w;
            bF[4]  = (_Float16)x1.x; bF[5]  = (_Float16)x1.y;
            bF[6]  = (_Float16)x1.z; bF[7]  = (_Float16)x1.w;
            bF[8]  = (_Float16)x2.x; bF[9]  = (_Float16)x2.y;
            bF[10] = (_Float16)x2.z; bF[11] = (_Float16)x2.w;
            bF[12] = (_Float16)x3.x; bF[13] = (_Float16)x3.y;
            bF[14] = (_Float16)x3.z; bF[15] = (_Float16)x3.w;
            acc = __builtin_amdgcn_wmma_f32_16x16x32_f16(
                false, aF[dc], false, bF, (short)0, acc, false, false);
        }
        // C layout: VGPR v -> row M = mrow + v, col N = lane&15.
        // Double-exp of the reference: e = exp(exp(qk)/sqrt(d)); mask applied later.
        #pragma unroll
        for (int v = 0; v < 8; ++v) {
            float e = __expf(__expf(acc[v]) * kInvSqrtD);
            sE[mrow + v][kv0 + bn] = (_Float16)e;
        }
    }
    __syncthreads();

    // ---- Masked row sums -> rinv (8 threads per row, 128-column segment each) ----
    {
        const int r   = tid >> 3;
        const int seg = tid & 7;
        const int* mrowp = mask + ((size_t)b * kSQ + q0 + r) * kSKV;
        float s = 0.0f;
        const int k0 = seg * (kSKV / 8);
        for (int k = k0; k < k0 + kSKV / 8; ++k)
            s += mrowp[k] ? (float)sE[r][k] : 0.0f;
        sRowPart[r][seg] = s;
    }
    __syncthreads();
    if (tid < kQTile) {
        float s = 0.0f;
        #pragma unroll
        for (int j = 0; j < 8; ++j) s += sRowPart[tid][j];
        sRinv[tid] = 1.0f / fmaxf(s, 1e-30f);
    }
    __syncthreads();

    // ---- Write p_attn (coalesced) and fold column weights w[k] = sum_m p[m][k] ----
    {
        const size_t pbase = ((size_t)b * kSQ + q0) * kSKV;
        const int*   mbase = mask + pbase;
        for (int k = tid; k < kSKV; k += kThreads) {
            float wk = 0.0f;
            #pragma unroll
            for (int m = 0; m < kQTile; ++m) {
                const int msk = mbase[(size_t)m * kSKV + k];
                float p = msk ? (float)sE[m][k] * sRinv[m] : 0.0f;
                pOut[pbase + (size_t)m * kSKV + k] = p;
                wk += p;
            }
            sW[k] = wk;
        }
    }
    __syncthreads();

    // ---- Weighted-V pass: out[d] += sum_k w[k] * V[k][d]  (sum over q folded) ----
    {
        const int d0 = tid * 2;                    // 128 threads x float2 = 256 cols
        const float2* v2 = reinterpret_cast<const float2*>(
            V + (size_t)b * kSKV * kD + d0);
        float ax = 0.0f, ay = 0.0f;
        for (int k = 0; k < kSKV; ++k) {
            const float w = sW[k];
            float2 vv = v2[(size_t)k * (kD / 2)];
            ax = fmaf(w, vv.x, ax);
            ay = fmaf(w, vv.y, ay);
        }
        atomicAdd(&outSum[b * kD + d0],     ax);
        atomicAdd(&outSum[b * kD + d0 + 1], ay);
    }
}

extern "C" void kernel_launch(void* const* d_in, const int* in_sizes, int n_in,
                              void* d_out, int out_size, void* d_ws, size_t ws_size,
                              hipStream_t stream) {
    const float* Q    = (const float*)d_in[0];
    const float* K    = (const float*)d_in[1];
    const float* V    = (const float*)d_in[2];
    const int*   mask = (const int*)d_in[3];

    float* outSum = (float*)d_out;                    // B*D = 4096 floats
    float* pOut   = (float*)d_out + (size_t)kB * kD;  // B*SQ*SKV floats

    // Harness poisons d_out; zero the atomic-accumulated region first.
    zero_outsum_kernel<<<kB * kD / 256, 256, 0, stream>>>(outSum);

    dim3 grid(kSQ / kQTile, kB);   // 64 x 16 workgroups
    attention_tile_kernel<<<grid, kThreads, 0, stream>>>(Q, K, V, mask, outSum, pOut);
}
